// LatentODE1_85718957293959
// MI455X (gfx1250) — matrix operand order repaired and checked by reading, not compiled, not run on hardware
//
#include <hip/hip_runtime.h>
#include <hip/hip_bf16.h>

typedef __attribute__((ext_vector_type(16))) _Float16 v16h;
typedef __attribute__((ext_vector_type(2)))  _Float16 h2v;
typedef __attribute__((ext_vector_type(8)))  float    v8f;

#define T_SEQ 100
#define IN_F  32
#define N_RK  8

// ---------- CDNA5 WMMA: D = A(16x32 f16) * B(32x16 f16) + C(16x16 f32) ----------
static __device__ __forceinline__ v8f wmma_f16(v16h a, v16h b, v8f c) {
  return __builtin_amdgcn_wmma_f32_16x16x32_f16(false, a, false, b, (short)0, c, false, false);
}

// ---------- hardware tanh (gfx1250 transcendental) ----------
static __device__ __forceinline__ float fast_tanh(float x) {
#if __has_builtin(__builtin_amdgcn_tanhf)
  return __builtin_amdgcn_tanhf(x);
#elif __has_builtin(__builtin_amdgcn_tanh_f32)
  return __builtin_amdgcn_tanh_f32(x);
#else
  float r;
  // TRANS op: 1 gap instruction required before result use (ISA 7.4 case 3)
  asm volatile("v_tanh_f32 %0, %1\n\tv_nop" : "=v"(r) : "v"(x));
  return r;
#endif
}

// Load W^T (from row-major W[Ksrc x Msrc]) as a WMMA A fragment (16x32, f16).
// Branchless: clamp the address so every load is valid, zero via select.
static __device__ __forceinline__ v16h load_wT_afrag(const float* __restrict__ W,
                                                     int Ksrc, int Msrc,
                                                     int mtile, int kchunk, int lane) {
  const int m    = (lane & 15) + (mtile << 4);
  const int kadd = ((lane & 16) ? 16 : 0) + (kchunk << 5);
  const int mc   = (m < Msrc) ? m : 0;
  v16h a;
#pragma unroll
  for (int e = 0; e < 16; ++e) {
    const int k  = e + kadd;
    const int kc = (k < Ksrc) ? k : 0;
    float v = W[kc * Msrc + mc];
    v = (m < Msrc && k < Ksrc) ? v : 0.0f;
    a[e] = (_Float16)v;
  }
  return a;
}

// Wf2^T fragment with bf2 folded in as K=20 row (H row 20 is forced to 1.0).
static __device__ __forceinline__ v16h load_wf2_afrag(const float* __restrict__ Wf2,
                                                      const float* __restrict__ bf2,
                                                      int mtile, int lane) {
  const int m    = (lane & 15) + (mtile << 4);   // 0..31, always valid
  const int kadd = (lane & 16) ? 16 : 0;
  const float bv = bf2[m];
  v16h a;
#pragma unroll
  for (int e = 0; e < 16; ++e) {
    const int k  = e + kadd; // 0..31
    const int kc = (k < 20) ? k : 0;
    float v = Wf2[kc * 32 + m];
    v = (k < 20) ? v : ((k == 20) ? bv : 0.0f);
    a[e] = (_Float16)v;
  }
  return a;
}

// Bias replicated into C-layout: lane half selects M = v + 8*(lane>=16) + 16*mtile
static __device__ __forceinline__ void load_bias_c(const float* __restrict__ b, int len,
                                                   int mtile, int lane, float o[8]) {
  const int mb = (mtile << 4) + ((lane & 16) ? 8 : 0);
#pragma unroll
  for (int v = 0; v < 8; ++v) {
    const int m  = mb + v;
    const int mc = (m < len) ? m : 0;
    float bv = b[mc];
    o[v] = (m < len) ? bv : 0.0f;
  }
}

// Convert two 16-row C tiles (f32) into the next GEMM's f16 B fragment.
// Pack K-pairs to f16 first, then swap lane halves with ONE shfl per packed reg:
// 8 cvt_pk + 8 shfl_xor(16) + 8 selects total.
static __device__ __forceinline__ v16h c_to_bfrag(const v8f c0, const v8f c1, int lane) {
  const bool hi = (lane & 16) != 0;
  unsigned p0[4], p1[4];
#pragma unroll
  for (int j = 0; j < 4; ++j) {
    h2v a; a.x = (_Float16)c0[2 * j]; a.y = (_Float16)c0[2 * j + 1];
    h2v b; b.x = (_Float16)c1[2 * j]; b.y = (_Float16)c1[2 * j + 1];
    p0[j] = __builtin_bit_cast(unsigned, a);
    p1[j] = __builtin_bit_cast(unsigned, b);
  }
  v16h r;
#pragma unroll
  for (int j = 0; j < 4; ++j) {
    const unsigned q0 = (unsigned)__shfl_xor((int)p0[j], 16, 32);
    const unsigned q1 = (unsigned)__shfl_xor((int)p1[j], 16, 32);
    // lanes<16: K 0..7 = local p0, K 8..15 = remote p0
    // lanes>=16: K16..23 = remote p1, K24..31 = local p1
    const h2v lo = __builtin_bit_cast(h2v, hi ? q1 : p0[j]);
    const h2v hh = __builtin_bit_cast(h2v, hi ? p1[j] : q0);
    r[2 * j]     = lo.x; r[2 * j + 1]     = lo.y;
    r[8 + 2 * j] = hh.x; r[8 + 2 * j + 1] = hh.y;
  }
  return r;
}

// x[b0+n, t, :] as B fragment: lane n(<16) holds K=0..15, lane n+16 holds K=16..31
static __device__ __forceinline__ v16h load_x_bfrag(const float* __restrict__ x,
                                                    int b0, int t, int lane) {
  const int n = lane & 15;
  const size_t off = ((size_t)(b0 + n) * T_SEQ + t) * IN_F + ((lane & 16) ? 16 : 0);
  const float4* q = (const float4*)(x + off);
  const float4 a = q[0], b = q[1], c = q[2], d = q[3];
  const float f[16] = {a.x, a.y, a.z, a.w, b.x, b.y, b.z, b.w,
                       c.x, c.y, c.z, c.w, d.x, d.y, d.z, d.w};
  v16h r;
#pragma unroll
  for (int e = 0; e < 16; ++e) r[e] = (_Float16)f[e];
  return r;
}

struct FNet {
  v16h wf1[2][2];   // [m-tile][k-chunk: z|x]  Wf1^T fragments
  v16h wf2[2];      // [m-tile] Wf2^T fragments (K pad 20->32, bf2 folded at K=20)
  float bf1[2][8];  // bf1 padded; entry for m==20 is 1.0 -> H row 20 == 1
};

// k = tanh(Wf2ext^T * relu(Wf1^T * [z;x] + bf1pad)) * dscale   (C layout out)
static __device__ __forceinline__ void f_eval(const FNet& w, v16h zB, v16h xB,
                                              float dscale, int lane,
                                              v8f& k0, v8f& k1) {
  v8f h0 = {}; v8f h1 = {};
  h0 = wmma_f16(w.wf1[0][0], zB, h0);
  h0 = wmma_f16(w.wf1[0][1], xB, h0);
  h1 = wmma_f16(w.wf1[1][0], zB, h1);
  h1 = wmma_f16(w.wf1[1][1], xB, h1);
#pragma unroll
  for (int v = 0; v < 8; ++v) {
    h0[v] = fmaxf(h0[v] + w.bf1[0][v], 0.0f);
    h1[v] = fmaxf(h1[v] + w.bf1[1][v], 0.0f);
  }
  const v16h hB = c_to_bfrag(h0, h1, lane);
  v8f o0 = {}; v8f o1 = {};
  o0 = wmma_f16(w.wf2[0], hB, o0);   // includes bf2 via constant-1 H row
  o1 = wmma_f16(w.wf2[1], hB, o1);
#pragma unroll
  for (int v = 0; v < 8; ++v) {
    k0[v] = fast_tanh(o0[v]) * dscale;
    k1[v] = fast_tanh(o1[v]) * dscale;
  }
}

__global__ void __launch_bounds__(256)
latent_ode_kernel(const float* __restrict__ dt, const float* __restrict__ x,
                  const float* __restrict__ We1, const float* __restrict__ be1,
                  const float* __restrict__ We2, const float* __restrict__ be2,
                  const float* __restrict__ Wf1, const float* __restrict__ bf1,
                  const float* __restrict__ Wf2, const float* __restrict__ bf2,
                  const float* __restrict__ Wm1, const float* __restrict__ bm1,
                  const float* __restrict__ Wm2, const float* __restrict__ bm2,
                  float* __restrict__ out, int B) {
  const int lane = threadIdx.x & 31;
  const int wave = threadIdx.x >> 5;
  const int tile = blockIdx.x * (blockDim.x >> 5) + wave;
  if (tile >= (B >> 4)) return;         // wave-uniform exit; WMMA needs EXEC all-1
  const int b0 = tile << 4;

  // ---- persistent weight fragments (registers; shared weights stay in L2) ----
  FNet w;
  w.wf1[0][0] = load_wT_afrag(Wf1, 64, 20, 0, 0, lane);  // z chunk, m 0..15
  w.wf1[0][1] = load_wT_afrag(Wf1, 64, 20, 0, 1, lane);  // x chunk
  w.wf1[1][0] = load_wT_afrag(Wf1, 64, 20, 1, 0, lane);  // m 16..31 (pad >=20)
  w.wf1[1][1] = load_wT_afrag(Wf1, 64, 20, 1, 1, lane);
  w.wf2[0]    = load_wf2_afrag(Wf2, bf2, 0, lane);
  w.wf2[1]    = load_wf2_afrag(Wf2, bf2, 1, lane);
  load_bias_c(bf1, 20, 0, lane, w.bf1[0]);
  // tile1 bias: m 16..19 from bf1, m==20 forced to 1.0 (bias-fold row), else 0
  {
    const int mb = 16 + ((lane & 16) ? 8 : 0);
#pragma unroll
    for (int v = 0; v < 8; ++v) {
      const int m  = mb + v;
      const int mc = (m < 20) ? m : 0;
      float bv = bf1[mc];
      w.bf1[1][v] = (m < 20) ? bv : ((m == 20) ? 1.0f : 0.0f);
    }
  }

  const v16h wm1f = load_wT_afrag(Wm1, 32, 10, 0, 0, lane);
  float bm1l[8]; load_bias_c(bm1, 10, 0, lane, bm1l);
  float wm2l[8];
  {
    const int mb = (lane & 16) ? 8 : 0;
#pragma unroll
    for (int v = 0; v < 8; ++v) {
      const int m  = mb + v;
      const int mc = (m < 10) ? m : 0;
      float wv = Wm2[mc];
      wm2l[v] = (m < 10) ? wv : 0.0f;
    }
  }
  const float bm2v = bm2[0];

  // ---- encoder: z0 = tanh(We2^T * tanh(We1^T * x0 + be1) + be2) ----
  v16h xB = load_x_bfrag(x, b0, 0, lane);
  v8f z0c, z1c;
  {
    const v16h we1f = load_wT_afrag(We1, 32, 10, 0, 0, lane);
    v8f e = {};
    e = wmma_f16(we1f, xB, e);
    float be1l[8]; load_bias_c(be1, 10, 0, lane, be1l);
#pragma unroll
    for (int v = 0; v < 8; ++v) e[v] = fast_tanh(e[v] + be1l[v]);
    const v8f zz = {};
    const v16h eB = c_to_bfrag(e, zz, lane);          // K pad 10->32 (zeros)
    const v16h we2t0 = load_wT_afrag(We2, 10, 32, 0, 0, lane);
    const v16h we2t1 = load_wT_afrag(We2, 10, 32, 1, 0, lane);
    z0c = v8f{}; z1c = v8f{};
    z0c = wmma_f16(we2t0, eB, z0c);
    z1c = wmma_f16(we2t1, eB, z1c);
    float b20[8], b21[8];
    load_bias_c(be2, 32, 0, lane, b20);
    load_bias_c(be2, 32, 1, lane, b21);
#pragma unroll
    for (int v = 0; v < 8; ++v) {
      z0c[v] = fast_tanh(z0c[v] + b20[v]);
      z1c[v] = fast_tanh(z1c[v] + b21[v]);
    }
  }

  v16h zB = c_to_bfrag(z0c, z1c, lane);     // invariant: zB matches (z0c,z1c)
  const float hs = 1.0f / (float)N_RK;
  const float h2 = 0.5f * hs;
  const float h6 = hs / 6.0f;

  for (int t = 0; t < T_SEQ; ++t) {
    if (t > 0) xB = load_x_bfrag(x, b0, t, lane);
    if (t + 1 < T_SEQ) {                    // gfx1250 global_prefetch_b8
      const int n = lane & 15;
      const size_t poff = ((size_t)(b0 + n) * T_SEQ + (t + 1)) * IN_F + ((lane & 16) ? 16 : 0);
      __builtin_prefetch(x + poff, 0, 3);
    }
    float dsc;
    {
      const int n = lane & 15;
      const float2 dd = *(const float2*)(dt + ((size_t)(b0 + n) * T_SEQ + t) * 2);
      dsc = (dd.y - dd.x) * 0.01f;          // per-column (batch) scale, DT_SCALER
    }

#pragma unroll 1
    for (int s = 0; s < N_RK; ++s) {
      v8f ka, kb, acc0, acc1, t0, t1;
      f_eval(w, zB, xB, dsc, lane, ka, kb);                 // k1
      acc0 = ka; acc1 = kb;
#pragma unroll
      for (int v = 0; v < 8; ++v) { t0[v] = z0c[v] + h2 * ka[v]; t1[v] = z1c[v] + h2 * kb[v]; }
      v16h tB = c_to_bfrag(t0, t1, lane);
      f_eval(w, tB, xB, dsc, lane, ka, kb);                 // k2
#pragma unroll
      for (int v = 0; v < 8; ++v) {
        acc0[v] += 2.0f * ka[v]; acc1[v] += 2.0f * kb[v];
        t0[v] = z0c[v] + h2 * ka[v]; t1[v] = z1c[v] + h2 * kb[v];
      }
      tB = c_to_bfrag(t0, t1, lane);
      f_eval(w, tB, xB, dsc, lane, ka, kb);                 // k3
#pragma unroll
      for (int v = 0; v < 8; ++v) {
        acc0[v] += 2.0f * ka[v]; acc1[v] += 2.0f * kb[v];
        t0[v] = z0c[v] + hs * ka[v]; t1[v] = z1c[v] + hs * kb[v];
      }
      tB = c_to_bfrag(t0, t1, lane);
      f_eval(w, tB, xB, dsc, lane, ka, kb);                 // k4
#pragma unroll
      for (int v = 0; v < 8; ++v) {
        z0c[v] += h6 * (acc0[v] + ka[v]);
        z1c[v] += h6 * (acc1[v] + kb[v]);
      }
      zB = c_to_bfrag(z0c, z1c, lane);      // feeds next eval AND mu head
    }

    // ---- mu = Wm2^T * tanh(Wm1^T * z + bm1) + bm2 ----
    v8f g = {};
    g = wmma_f16(wm1f, zB, g);
    float part = 0.0f;
#pragma unroll
    for (int v = 0; v < 8; ++v) part += wm2l[v] * fast_tanh(g[v] + bm1l[v]);
    part += __shfl_xor(part, 16, 32);       // combine M halves
    const float mu = part + bm2v;
    if (lane < 16) out[(size_t)(b0 + lane) * T_SEQ + t] = mu;
  }
}

extern "C" void kernel_launch(void* const* d_in, const int* in_sizes, int n_in,
                              void* d_out, int out_size, void* d_ws, size_t ws_size,
                              hipStream_t stream) {
  (void)n_in; (void)out_size; (void)d_ws; (void)ws_size;
  const float* dt  = (const float*)d_in[0];
  const float* x   = (const float*)d_in[1];
  const float* We1 = (const float*)d_in[2];
  const float* be1 = (const float*)d_in[3];
  const float* We2 = (const float*)d_in[4];
  const float* be2 = (const float*)d_in[5];
  const float* Wf1 = (const float*)d_in[6];
  const float* bf1 = (const float*)d_in[7];
  const float* Wf2 = (const float*)d_in[8];
  const float* bf2 = (const float*)d_in[9];
  const float* Wm1 = (const float*)d_in[10];
  const float* bm1 = (const float*)d_in[11];
  const float* Wm2 = (const float*)d_in[12];
  const float* bm2 = (const float*)d_in[13];
  float* out = (float*)d_out;

  const int B = in_sizes[1] / (T_SEQ * IN_F);   // x is [B, T, 32]
  const int tiles = B / 16;                     // 16 batch rows per wave
  const int wavesPerBlock = 8;                  // 256 threads = 8 wave32
  const int blocks = (tiles + wavesPerBlock - 1) / wavesPerBlock;
  latent_ode_kernel<<<blocks, wavesPerBlock * 32, 0, stream>>>(
      dt, x, We1, be1, We2, be2, Wf1, bf1, Wf2, bf2, Wm1, bm1, Wm2, bm2, out, B);
}
